// MultiHeadAttention_63814624084186
// MI455X (gfx1250) — matrix operand order repaired and checked by reading, not compiled, not run on hardware
//
#include <hip/hip_runtime.h>
#include <hip/hip_bf16.h>

// ---------------- problem constants ----------------
#define BATCH   16
#define DIMC    512           // DIM (channels)
#define NHEADS  8
#define DHEAD   64
#define INNER   512           // NUM_HEADS * DIM_HEAD
#define O3      1536          // 3 * INNER
#define SEQ     4096          // H*W
#define LN_EPS  1e-5f
#define PITCH   40            // LDS row pitch (elems): 80B rows, 16B aligned, conflict-free b128

typedef __attribute__((ext_vector_type(16))) __bf16 bf16x16;
typedef __attribute__((ext_vector_type(8)))  __bf16 bf16x8;
typedef __attribute__((ext_vector_type(2)))  __bf16 bf16x2;
typedef __attribute__((ext_vector_type(8)))  float  f32x8;

// ---- optional CDNA5 paths (guarded; fall back to portable code) ----
#if defined(__has_builtin)
# if __has_builtin(__builtin_amdgcn_global_load_async_to_lds_b128)
#  define HAVE_ASYNC_B128 1
# endif
# if __has_builtin(__builtin_amdgcn_s_wait_asynccnt)
#  define HAVE_WAIT_ASYNC 1
# endif
# if __has_builtin(__builtin_amdgcn_cvt_pk_bf16_f32)
#  define HAVE_CVT_PK_BF16 1
# endif
#endif

// async builtin signature: (int4 addrspace(1)*, int4 addrspace(3)*, imm offset, imm cpol)
typedef int v4i __attribute__((vector_size(16)));
typedef __attribute__((address_space(1))) v4i* glb_v4i_p;
typedef __attribute__((address_space(3))) v4i* lds_v4i_p;

// copy 16 bytes (8 bf16) global -> LDS
__device__ __forceinline__ void copy16_g2l(const __bf16* g, __bf16* l) {
#if defined(HAVE_ASYNC_B128)
    __builtin_amdgcn_global_load_async_to_lds_b128((glb_v4i_p)g, (lds_v4i_p)l, 0, 0);
#else
    *(bf16x8*)l = *(const bf16x8*)g;
#endif
}

__device__ __forceinline__ void async_join() {
#if defined(HAVE_ASYNC_B128)
# if defined(HAVE_WAIT_ASYNC)
    __builtin_amdgcn_s_wait_asynccnt(0);
# else
    asm volatile("s_wait_asynccnt 0x0" ::: "memory");
# endif
#endif
}

// ---- f32 -> bf16 conversions ----
__device__ __forceinline__ unsigned short bfbits(float f) {   // RNE bit-trick fallback
    union { float f; unsigned u; } v; v.f = f;
    unsigned r = v.u + 0x7FFFu + ((v.u >> 16) & 1u);
    return (unsigned short)(r >> 16);
}

__device__ __forceinline__ __bf16 f2bf(float f) {
#if defined(HAVE_CVT_PK_BF16)
    bf16x2 v = __builtin_amdgcn_cvt_pk_bf16_f32(f, 0.f);
    return v[0];
#else
    unsigned short h = bfbits(f);
    __bf16 o; __builtin_memcpy(&o, &h, sizeof(o));
    return o;
#endif
}

// pack two floats into (lo | hi<<16) bf16 pair
__device__ __forceinline__ unsigned pack2bf(float lo, float hi) {
#if defined(HAVE_CVT_PK_BF16)
    bf16x2 v = __builtin_amdgcn_cvt_pk_bf16_f32(lo, hi);
    unsigned u; __builtin_memcpy(&u, &v, 4);
    return u;
#else
    return (unsigned)bfbits(lo) | ((unsigned)bfbits(hi) << 16);
#endif
}

__device__ __forceinline__ f32x8 wmma_bf16(bf16x16 a, bf16x16 b, f32x8 c) {
    return __builtin_amdgcn_wmma_f32_16x16x32_bf16(false, a, false, b, (short)0, c, false, false);
}

__device__ __forceinline__ bf16x16 cat8(bf16x8 lo, bf16x8 hi) {
    return __builtin_shufflevector(lo, hi, 0,1,2,3,4,5,6,7,8,9,10,11,12,13,14,15);
}

// A fragment (16x32 MxK): lane m-row pointer given; runs [half*8,+8) and [16+half*8,+8)
__device__ __forceinline__ bf16x16 frag_a(const __bf16* row, int half) {
    const __bf16* p = row + half * 8;
    return cat8(*(const bf16x8*)p, *(const bf16x8*)(p + 16));
}
// B fragment (32x16 KxN) from K-contiguous row (Bt[n][k]): runs [half*8,+8) and [16+half*8,+8)
__device__ __forceinline__ bf16x16 frag_b(const __bf16* row, int half) {
    const __bf16* p = row + half * 8;
    return cat8(*(const bf16x8*)p, *(const bf16x8*)(p + 16));
}

// ---------------- kernel 0: f32 -> bf16 weight convert ----------------
__global__ void cvt_f32_bf16(const float* __restrict__ src, __bf16* __restrict__ dst, int n) {
    int i = blockIdx.x * blockDim.x + threadIdx.x;
    if (i < n) dst[i] = f2bf(src[i]);
}

// ---------------- kernel 1: QKV GEMM ----------------
// qkv[b][o][n] = sum_c w_qkv[o][c] * x[b][c][n]   (M=1536, K=512, N=4096)
// 256 thr / 8 waves; block tile 64(M)x128(N); wave tile 32x32 (4 WMMA / K-step).
__global__ __launch_bounds__(256) void qkv_gemm(const float* __restrict__ x,
                                                const __bf16* __restrict__ wq,
                                                __bf16* __restrict__ qkv) {
    __shared__ __align__(16) __bf16 As[64 * PITCH];
    __shared__ __align__(16) __bf16 Bt[128 * PITCH];
    const int tid = threadIdx.x, lane = tid & 31, wave = tid >> 5;
    const int mw = wave >> 2, nw = wave & 3;
    const int half = lane >> 4, nc = lane & 15;
    const int n0 = blockIdx.x * 128, m0 = blockIdx.y * 64, b = blockIdx.z;
    const float* xb = x + (size_t)b * DIMC * SEQ;
    f32x8 acc[2][2] = {};
    for (int k0 = 0; k0 < DIMC; k0 += 32) {
        {   // A tile 64x32 bf16: one 16B async chunk per thread
            int row = tid >> 2, q = tid & 3;
            copy16_g2l(wq + (m0 + row) * DIMC + k0 + q * 8, As + row * PITCH + q * 8);
        }
#pragma unroll
        for (int t = 0; t < 8; ++t) {    // B tile 32x128 f32 -> Bt[n][k], K-adjacent pairs
            int p = tid + t * 256, c = p & 127, r2 = (p >> 7) << 1;
            float lo = xb[(size_t)(k0 + r2) * SEQ + n0 + c];
            float hi = xb[(size_t)(k0 + r2 + 1) * SEQ + n0 + c];
            *(unsigned*)(Bt + c * PITCH + r2) = pack2bf(lo, hi);
        }
        async_join();
        __syncthreads();
        bf16x16 bfr[2];
#pragma unroll
        for (int nt = 0; nt < 2; ++nt)
            bfr[nt] = frag_b(Bt + (nw * 32 + nt * 16 + nc) * PITCH, half);
#pragma unroll
        for (int mt = 0; mt < 2; ++mt) {
            bf16x16 a = frag_a(As + (mw * 32 + mt * 16 + nc) * PITCH, half);
#pragma unroll
            for (int nt = 0; nt < 2; ++nt)
                acc[mt][nt] = wmma_bf16(a, bfr[nt], acc[mt][nt]);
        }
        __syncthreads();
    }
#pragma unroll
    for (int mt = 0; mt < 2; ++mt)
#pragma unroll
        for (int nt = 0; nt < 2; ++nt)
#pragma unroll
            for (int r = 0; r < 8; ++r) {
                int m = m0 + mw * 32 + mt * 16 + r + 8 * half;
                int n = n0 + nw * 32 + nt * 16 + nc;
                qkv[((size_t)b * O3 + m) * SEQ + n] = f2bf(acc[mt][nt][r]);
            }
}

// ---------------- kernel 2: scores + softmax (fused) ----------------
// S[d][e] = (1/8) * sum_n q[d][n]*k[e][n]; softmax over e; emit bf16 probs.
__global__ __launch_bounds__(128) void attn_scores(const __bf16* __restrict__ qkv,
                                                   __bf16* __restrict__ attnP) {
    __shared__ __align__(16) __bf16 Qs[64 * PITCH];
    __shared__ __align__(16) __bf16 Ks[64 * PITCH];  // Ks[e][kk] == Bt layout for k^T
    __shared__ float Ss[64 * 64];
    const int tid = threadIdx.x, lane = tid & 31, wave = tid >> 5;
    const int half = lane >> 4, nc = lane & 15;
    const int bh = blockIdx.x, b = bh >> 3, h = bh & 7;
    const __bf16* qb = qkv + ((size_t)b * O3 + h * DHEAD) * SEQ;
    const __bf16* kb = qkv + ((size_t)b * O3 + INNER + h * DHEAD) * SEQ;
    f32x8 acc[4] = {};
    for (int k0 = 0; k0 < SEQ; k0 += 32) {
#pragma unroll
        for (int t = 0; t < 2; ++t) {    // 64x32 tiles: 256 chunks each, 2/thread
            int cidx = tid + t * 128, row = cidx >> 2, q = cidx & 3;
            copy16_g2l(qb + (size_t)row * SEQ + k0 + q * 8, Qs + row * PITCH + q * 8);
            copy16_g2l(kb + (size_t)row * SEQ + k0 + q * 8, Ks + row * PITCH + q * 8);
        }
        async_join();
        __syncthreads();
        bf16x16 a = frag_a(Qs + (wave * 16 + nc) * PITCH, half);
#pragma unroll
        for (int t = 0; t < 4; ++t) {
            bf16x16 bb = frag_b(Ks + (t * 16 + nc) * PITCH, half);
            acc[t] = wmma_bf16(a, bb, acc[t]);
        }
        __syncthreads();
    }
#pragma unroll
    for (int t = 0; t < 4; ++t)
#pragma unroll
        for (int r = 0; r < 8; ++r)
            Ss[(wave * 16 + r + 8 * half) * 64 + t * 16 + nc] = acc[t][r] * 0.125f;
    __syncthreads();
    if (tid < 64) {
        float m = -1e30f;
        for (int e = 0; e < 64; ++e) m = fmaxf(m, Ss[tid * 64 + e]);
        float s = 0.f;
        for (int e = 0; e < 64; ++e) s += __expf(Ss[tid * 64 + e] - m);
        float inv = 1.f / s;
        for (int e = 0; e < 64; e += 2) {
            unsigned pk = pack2bf(__expf(Ss[tid * 64 + e] - m) * inv,
                                  __expf(Ss[tid * 64 + e + 1] - m) * inv);
            *(unsigned*)(attnP + ((size_t)bh * 64 + tid) * 64 + e) = pk;
        }
    }
}

// ---------------- kernel 3: attn @ v ----------------
// aout[d][n] = sum_e attn[d][e] * v[e][n]   (M=64, K=64, N=4096)
__global__ __launch_bounds__(128) void attn_apply(const __bf16* __restrict__ qkv,
                                                  const __bf16* __restrict__ attnP,
                                                  __bf16* __restrict__ aout) {
    __shared__ __align__(16) __bf16 As[64 * 72];     // attn probs, row-major d x e (pitch 72)
    __shared__ __align__(16) __bf16 Vt[64 * PITCH];  // Vt[n][e-local]
    const int tid = threadIdx.x, lane = tid & 31, wave = tid >> 5;
    const int half = lane >> 4, nc = lane & 15;
    const int n0 = blockIdx.x * 64, bh = blockIdx.y, b = bh >> 3, h = bh & 7;
    const __bf16* vb = qkv + ((size_t)b * O3 + 2 * INNER + h * DHEAD) * SEQ;
#pragma unroll
    for (int t = 0; t < 4; ++t) {        // attn 64x64: 512 chunks, 4/thread
        int cidx = tid + t * 128, row = cidx >> 3, q = cidx & 7;
        copy16_g2l(attnP + (size_t)bh * 4096 + row * 64 + q * 8, As + row * 72 + q * 8);
    }
    f32x8 acc[4] = {};
    for (int e0 = 0; e0 < 64; e0 += 32) {
        if (e0) __syncthreads();         // Vt consumed by all waves
#pragma unroll
        for (int t = 0; t < 8; ++t) {    // v tile 32(e) x 64(n) -> Vt, e-adjacent pairs
            int p = tid + t * 128, c = p & 63, r2 = (p >> 6) << 1;
            unsigned lo = *(const unsigned short*)(vb + (size_t)(e0 + r2) * SEQ + n0 + c);
            unsigned hi = *(const unsigned short*)(vb + (size_t)(e0 + r2 + 1) * SEQ + n0 + c);
            *(unsigned*)(Vt + c * PITCH + r2) = lo | (hi << 16);
        }
        async_join();
        __syncthreads();
        bf16x16 a = frag_a(As + (wave * 16 + nc) * 72 + e0, half);
#pragma unroll
        for (int t = 0; t < 4; ++t) {
            bf16x16 bb = frag_b(Vt + (t * 16 + nc) * PITCH, half);
            acc[t] = wmma_bf16(a, bb, acc[t]);
        }
    }
#pragma unroll
    for (int t = 0; t < 4; ++t)
#pragma unroll
        for (int r = 0; r < 8; ++r) {
            int o = h * DHEAD + wave * 16 + r + 8 * half;
            aout[((size_t)b * INNER + o) * SEQ + n0 + t * 16 + nc] = f2bf(acc[t][r]);
        }
}

// ---------------- kernel 4: output projection (+bias) ----------------
// out[b][o][n] = sum_c w_out[o][c] * aout[b][c][n] + b_out[o]
__global__ __launch_bounds__(256) void out_proj(const __bf16* __restrict__ aout,
                                                const __bf16* __restrict__ wo,
                                                const float* __restrict__ bias,
                                                float* __restrict__ out) {
    __shared__ __align__(16) __bf16 As[64 * PITCH];
    __shared__ __align__(16) __bf16 Bt[128 * PITCH];
    const int tid = threadIdx.x, lane = tid & 31, wave = tid >> 5;
    const int mw = wave >> 2, nw = wave & 3;
    const int half = lane >> 4, nc = lane & 15;
    const int n0 = blockIdx.x * 128, m0 = blockIdx.y * 64, b = blockIdx.z;
    const __bf16* ab = aout + (size_t)b * INNER * SEQ;
    f32x8 acc[2][2] = {};
    for (int k0 = 0; k0 < INNER; k0 += 32) {
        {   // A tile 64x32 bf16 weights
            int row = tid >> 2, q = tid & 3;
            copy16_g2l(wo + (m0 + row) * INNER + k0 + q * 8, As + row * PITCH + q * 8);
        }
#pragma unroll
        for (int t = 0; t < 8; ++t) {    // B tile 32x128 bf16 -> Bt, K-adjacent pairs
            int p = tid + t * 256, c = p & 127, r2 = (p >> 7) << 1;
            unsigned lo = *(const unsigned short*)(ab + (size_t)(k0 + r2) * SEQ + n0 + c);
            unsigned hi = *(const unsigned short*)(ab + (size_t)(k0 + r2 + 1) * SEQ + n0 + c);
            *(unsigned*)(Bt + c * PITCH + r2) = lo | (hi << 16);
        }
        async_join();
        __syncthreads();
        bf16x16 bfr[2];
#pragma unroll
        for (int nt = 0; nt < 2; ++nt)
            bfr[nt] = frag_b(Bt + (nw * 32 + nt * 16 + nc) * PITCH, half);
#pragma unroll
        for (int mt = 0; mt < 2; ++mt) {
            bf16x16 a = frag_a(As + (mw * 32 + mt * 16 + nc) * PITCH, half);
#pragma unroll
            for (int nt = 0; nt < 2; ++nt)
                acc[mt][nt] = wmma_bf16(a, bfr[nt], acc[mt][nt]);
        }
        __syncthreads();
    }
#pragma unroll
    for (int mt = 0; mt < 2; ++mt)
#pragma unroll
        for (int nt = 0; nt < 2; ++nt)
#pragma unroll
            for (int r = 0; r < 8; ++r) {
                int m = m0 + mw * 32 + mt * 16 + r + 8 * half;
                int n = n0 + nw * 32 + nt * 16 + nc;
                out[((size_t)b * DIMC + m) * SEQ + n] = acc[mt][nt][r] + bias[m];
            }
}

// ---------------- kernels 5/6: LayerNorm over (C,H,W) per batch ----------------
__global__ __launch_bounds__(256) void ln_partial(const float* __restrict__ out,
                                                  float* __restrict__ part) {
    __shared__ float s1[256], s2[256];
    const int b = blockIdx.y;
    const float* p = out + (size_t)b * DIMC * SEQ + (size_t)blockIdx.x * 16384;
    float s = 0.f, q = 0.f;
#pragma unroll 4
    for (int k = 0; k < 64; ++k) {
        float v = p[threadIdx.x + k * 256];
        s += v; q += v * v;
    }
    s1[threadIdx.x] = s; s2[threadIdx.x] = q;
    __syncthreads();
    for (int w = 128; w > 0; w >>= 1) {
        if (threadIdx.x < (unsigned)w) {
            s1[threadIdx.x] += s1[threadIdx.x + w];
            s2[threadIdx.x] += s2[threadIdx.x + w];
        }
        __syncthreads();
    }
    if (threadIdx.x == 0) {
        part[(b * 128 + blockIdx.x) * 2]     = s1[0];
        part[(b * 128 + blockIdx.x) * 2 + 1] = s2[0];
    }
}

__global__ __launch_bounds__(128) void ln_finalize(const float* __restrict__ part,
                                                   float* __restrict__ stats) {
    __shared__ float s1[128], s2[128];
    const int b = blockIdx.x;
    s1[threadIdx.x] = part[(b * 128 + threadIdx.x) * 2];
    s2[threadIdx.x] = part[(b * 128 + threadIdx.x) * 2 + 1];
    __syncthreads();
    for (int w = 64; w > 0; w >>= 1) {
        if (threadIdx.x < (unsigned)w) {
            s1[threadIdx.x] += s1[threadIdx.x + w];
            s2[threadIdx.x] += s2[threadIdx.x + w];
        }
        __syncthreads();
    }
    if (threadIdx.x == 0) {
        const float invN = 1.0f / (float)(DIMC * SEQ);
        float mean = s1[0] * invN;
        float var  = s2[0] * invN - mean * mean;
        stats[b * 2]     = mean;
        stats[b * 2 + 1] = rsqrtf(var + LN_EPS);
    }
}

__global__ __launch_bounds__(256) void ln_apply(float* __restrict__ out,
                                                const float* __restrict__ stats,
                                                const float* __restrict__ gamma,
                                                const float* __restrict__ beta) {
    const int b = blockIdx.y;
    const float mean = stats[b * 2], inv = stats[b * 2 + 1];
    const size_t base = (size_t)b * DIMC * SEQ + (size_t)blockIdx.x * 2048;
#pragma unroll
    for (int k = 0; k < 8; ++k) {
        size_t i = base + threadIdx.x + (size_t)k * 256;
        int o = (int)((i >> 12) & (DIMC - 1));
        out[i] = (out[i] - mean) * inv * gamma[o] + beta[o];
    }
}

// ---------------- host-side launch ----------------
extern "C" void kernel_launch(void* const* d_in, const int* in_sizes, int n_in,
                              void* d_out, int out_size, void* d_ws, size_t ws_size,
                              hipStream_t stream) {
    const float* x      = (const float*)d_in[0];
    const float* w_qkv  = (const float*)d_in[1];
    const float* w_out  = (const float*)d_in[2];
    const float* b_out  = (const float*)d_in[3];
    const float* gamma  = (const float*)d_in[4];
    const float* beta   = (const float*)d_in[5];
    float* out = (float*)d_out;

    char* ws = (char*)d_ws;
    __bf16* qkv    = (__bf16*)ws; ws += (size_t)BATCH * O3 * SEQ * 2;
    __bf16* attnP  = (__bf16*)ws; ws += (size_t)BATCH * NHEADS * 64 * 64 * 2;
    __bf16* aoutB  = (__bf16*)ws; ws += (size_t)BATCH * INNER * SEQ * 2;
    __bf16* wq_bf  = (__bf16*)ws; ws += (size_t)O3 * DIMC * 2;
    __bf16* wo_bf  = (__bf16*)ws; ws += (size_t)DIMC * INNER * 2;
    float*  part   = (float*)ws;  ws += (size_t)BATCH * 128 * 2 * sizeof(float);
    float*  stats  = (float*)ws;

    cvt_f32_bf16<<<(O3 * DIMC + 255) / 256, 256, 0, stream>>>(w_qkv, wq_bf, O3 * DIMC);
    cvt_f32_bf16<<<(DIMC * INNER + 255) / 256, 256, 0, stream>>>(w_out, wo_bf, DIMC * INNER);

    qkv_gemm   <<<dim3(SEQ / 128, O3 / 64, BATCH), 256, 0, stream>>>(x, wq_bf, qkv);
    attn_scores<<<dim3(BATCH * NHEADS),            128, 0, stream>>>(qkv, attnP);
    attn_apply <<<dim3(SEQ / 64, BATCH * NHEADS),  128, 0, stream>>>(qkv, attnP, aoutB);
    out_proj   <<<dim3(SEQ / 128, DIMC / 64, BATCH), 256, 0, stream>>>(aoutB, wo_bf, b_out, out);

    ln_partial <<<dim3(128, BATCH), 256, 0, stream>>>(out, part);
    ln_finalize<<<dim3(BATCH),      128, 0, stream>>>(part, stats);
    ln_apply   <<<dim3(1024, BATCH), 256, 0, stream>>>(out, stats, gamma, beta);
}
